// SKANLinear_6949257084951
// MI455X (gfx1250) — matrix operand order repaired and checked by reading, not compiled, not run on hardware
//
#include <hip/hip_runtime.h>

// ---- CDNA5 (gfx1250) vector types ------------------------------------------
typedef __attribute__((ext_vector_type(16))) __bf16       v16bf;
typedef __attribute__((ext_vector_type(8)))  float        v8f;
typedef __attribute__((ext_vector_type(8)))  unsigned int v8u;
typedef __attribute__((ext_vector_type(4)))  float        f32x4;

#define K_IN   256
#define N_OUT  256
#define W_LD   257            // weight row stride (IN+1); col 256 = bias term
#define NBLK   512            // 16-row blocks over B=8192
#define GRID_X 256

// pack two f32 into one dword of bf16 (RNE via hardware bf16 cvt)
__device__ __forceinline__ unsigned pack_bf16(float a, float b) {
    unsigned short ua = __builtin_bit_cast(unsigned short, (__bf16)a);
    unsigned short ub = __builtin_bit_cast(unsigned short, (__bf16)b);
    return (unsigned)ua | ((unsigned)ub << 16);
}

__device__ __forceinline__ v16bf abs_bf16x16(v16bf v) {
    v8u u = __builtin_bit_cast(v8u, v);
#pragma unroll
    for (int i = 0; i < 8; ++i) u[i] &= 0x7fff7fffu;
    return __builtin_bit_cast(v16bf, u);
}

// out = 0.5*(X @ W[:,:256]^T + |X| @ |W[:,:256]|^T) + relu(W[:,256])
__global__ __launch_bounds__(256)
void skan_wmma_bf16_kernel(const float* __restrict__ x,
                           const float* __restrict__ wt,
                           float* __restrict__ out) {
    // A-operand staging, pre-swizzled: [kstep(8)][dword e2(8)][lane(32)] = 8 KB
    __shared__ unsigned lds_a[8 * 8 * 32];

    const int t    = threadIdx.x;      // 256 threads = 8 waves
    const int lane = t & 31;
    const int wv   = t >> 5;           // 8 waves cover N=256, 32 cols each
    const int cb   = wv * 32;          // column base for this wave
    const int ln   = lane & 15;
    const int lh   = lane >> 4;        // half-wave (operand layout split)

    // ---- resident B operands: W rows as bf16, 2 n-tiles x 8 k-steps --------
    // B k order used here (contiguous, same as A; not AMD's table order): lanes 0-15 hold K=k0..k0+15 (2/VGPR), lanes 16-31 K=k0+16..k0+31
    v16bf Bs[2][8];
    float bias[2];
#pragma unroll
    for (int j = 0; j < 2; ++j) {
        const int n = cb + 16 * j + ln;
        const float* wrow = wt + (size_t)n * W_LD;
        bias[j] = fmaxf(wrow[256], 0.0f);            // relu of the bias column
        const int kb2 = lh * 16;
#pragma unroll
        for (int ks = 0; ks < 8; ++ks) {
            const float* p = wrow + ks * 32 + kb2;   // 16 contiguous fp32
            f32x4 q0 = *(const f32x4*)(p + 0);
            f32x4 q1 = *(const f32x4*)(p + 4);
            f32x4 q2 = *(const f32x4*)(p + 8);
            f32x4 q3 = *(const f32x4*)(p + 12);
            v8u u;
            u[0] = pack_bf16(q0.x, q0.y); u[1] = pack_bf16(q0.z, q0.w);
            u[2] = pack_bf16(q1.x, q1.y); u[3] = pack_bf16(q1.z, q1.w);
            u[4] = pack_bf16(q2.x, q2.y); u[5] = pack_bf16(q2.z, q2.w);
            u[6] = pack_bf16(q3.x, q3.y); u[7] = pack_bf16(q3.z, q3.w);
            Bs[j][ks] = __builtin_bit_cast(v16bf, u);
        }
    }

    // producer mapping: thread t fills the full A chunk for (ks = t>>5, dest-lane = t&31)
    const int pm  = lane & 15;                 // row within the 16-row tile
    const int pkb = (lane >> 4) << 4;          // 0 or 16 (A-layout half split)
    const int pk0 = wv * 32 + pkb;             // thread's ks == wv here

    for (int blk = blockIdx.x; blk < NBLK; blk += gridDim.x) {
        const int row0 = blk * 16;

        // ---- stage X block (16 rows) fp32 -> bf16 into swizzled LDS --------
        // A k order used here (contiguous, same as B), per lane: elements 0..15 -> K=k0..k0+15
        {
            const float* src = x + (size_t)(row0 + pm) * K_IN + pk0;
            f32x4 q0 = *(const f32x4*)(src + 0);
            f32x4 q1 = *(const f32x4*)(src + 4);
            f32x4 q2 = *(const f32x4*)(src + 8);
            f32x4 q3 = *(const f32x4*)(src + 12);
            unsigned* dst = &lds_a[wv * 256 + lane];
            dst[0 * 32] = pack_bf16(q0.x, q0.y);
            dst[1 * 32] = pack_bf16(q0.z, q0.w);
            dst[2 * 32] = pack_bf16(q1.x, q1.y);
            dst[3 * 32] = pack_bf16(q1.z, q1.w);
            dst[4 * 32] = pack_bf16(q2.x, q2.y);
            dst[5 * 32] = pack_bf16(q2.z, q2.w);
            dst[6 * 32] = pack_bf16(q3.x, q3.y);
            dst[7 * 32] = pack_bf16(q3.z, q3.w);
        }
        __syncthreads();

        // prefetch the next block of x while WMMAs run
        const int nblk = blk + gridDim.x;
        if (nblk < NBLK) {
            const int pr = t & 15, pseg = t >> 4;    // 16 rows x 16 x 64B segments
            __builtin_prefetch(x + (size_t)(nblk * 16 + pr) * K_IN + pseg * 16, 0, 0);
        }

        // ---- WMMA main loop: both GEMMs chained into one accumulator -------
        v8f acc[2] = {};
#pragma unroll
        for (int ks = 0; ks < 8; ++ks) {
            const unsigned* ap = &lds_a[ks * 256 + lane];   // conflict-free
            v8u au;
#pragma unroll
            for (int e = 0; e < 8; ++e) au[e] = ap[e * 32];
            v16bf As = __builtin_bit_cast(v16bf, au);
            v16bf Aa = abs_bf16x16(As);
#pragma unroll
            for (int j = 0; j < 2; ++j) {
                v16bf Ba = abs_bf16x16(Bs[j][ks]);
                acc[j] = __builtin_amdgcn_wmma_f32_16x16x32_bf16(
                    false, As, false, Bs[j][ks], (short)0, acc[j], false, false);
                acc[j] = __builtin_amdgcn_wmma_f32_16x16x32_bf16(
                    false, Aa, false, Ba, (short)0, acc[j], false, false);
            }
        }

        // ---- epilogue: out = 0.5*acc + bias --------------------------------
        // C layout: VGPR r, lanes 0-15 -> row r, lanes 16-31 -> row r+8
#pragma unroll
        for (int j = 0; j < 2; ++j) {
            const int n = cb + 16 * j + ln;
            float* orow = out + (size_t)(row0 + 8 * lh) * N_OUT + n;
#pragma unroll
            for (int r = 0; r < 8; ++r) {
                orow[(size_t)r * N_OUT] = fmaf(0.5f, acc[j][r], bias[j]);
            }
        }
        __syncthreads();   // protect LDS before next block's staging
    }
}

extern "C" void kernel_launch(void* const* d_in, const int* in_sizes, int n_in,
                              void* d_out, int out_size, void* d_ws, size_t ws_size,
                              hipStream_t stream) {
    (void)in_sizes; (void)n_in; (void)out_size; (void)d_ws; (void)ws_size;
    const float* x  = (const float*)d_in[0];   // (8192, 256) f32
    const float* wt = (const float*)d_in[1];   // (256, 257) f32
    float* out = (float*)d_out;                // (8192, 256) f32
    skan_wmma_bf16_kernel<<<dim3(GRID_X), dim3(256), 0, stream>>>(x, wt, out);
}
